// EnhancedGNNEncoder_12249246728352
// MI455X (gfx1250) — matrix operand order repaired and checked by reading, not compiled, not run on hardware
//
#include <hip/hip_runtime.h>
#include <hip/hip_fp16.h>

// ---------------------------------------------------------------------------
// Types for CDNA5 WMMA (wave32): A/B = v16h (16 f16 across 8 VGPRs),
// C/D = v8f (16x16 f32 tile, 8 VGPRs).
// ---------------------------------------------------------------------------
typedef __attribute__((ext_vector_type(16))) _Float16 v16h;
typedef __attribute__((ext_vector_type(8)))  _Float16 v8h;
typedef __attribute__((ext_vector_type(8)))  float    v8f;

#define SHUF16(lo, hi) __builtin_shufflevector((lo), (hi), \
    0,1,2,3,4,5,6,7,8,9,10,11,12,13,14,15)

static __device__ __forceinline__ v8f wmma_f16(v16h a, v16h b, v8f c) {
  return __builtin_amdgcn_wmma_f32_16x16x32_f16(
      /*neg_a=*/false, a, /*neg_b=*/false, b,
      /*c_mod=*/(short)0, c, /*reuse_a=*/false, /*reuse_b=*/false);
}

// Order-preserving float<->uint encoding so unsigned atomicMax == float max.
static __device__ __forceinline__ unsigned ford_enc(float f) {
  unsigned u = __float_as_uint(f);
  return (u & 0x80000000u) ? ~u : (u | 0x80000000u);
}
static __device__ __forceinline__ float ford_dec(unsigned u) {
  unsigned v = (u & 0x80000000u) ? (u & 0x7FFFFFFFu) : ~u;
  return __uint_as_float(v);
}

static constexpr int NN   = 4096;
static constexpr int NE   = 131072;
static constexpr int ETOT = NE + NN;   // self-loops appended

// ---------------------------------------------------------------------------
// Elementwise fp32 -> f16
// ---------------------------------------------------------------------------
__global__ void k_to_f16(const float* __restrict__ src, _Float16* __restrict__ dst, int n) {
  int t = blockIdx.x * blockDim.x + threadIdx.x;
  if (t < n) dst[t] = (_Float16)src[t];
}

// W [K x N] fp32 row-major -> Wt [N x K] f16 (B-operand wants N rows of contiguous K)
__global__ void k_transpose_f16(const float* __restrict__ W, _Float16* __restrict__ Wt,
                                int K, int N) {
  int t = blockIdx.x * blockDim.x + threadIdx.x;
  if (t >= K * N) return;
  int n = t / K, k = t - n * K;
  Wt[t] = (_Float16)W[(size_t)k * N + n];
}

// ---------------------------------------------------------------------------
// WMMA GEMM: C[MxN] = A[MxK](f16,row-major) * Bt[NxK](f16) + bias[N]
// 256 threads = 8 waves, each wave one 16x16 tile -> 64x32 block tile.
// ---------------------------------------------------------------------------
__global__ __launch_bounds__(256) void k_gemm_wmma(
    const _Float16* __restrict__ A, const _Float16* __restrict__ Bt,
    const float* __restrict__ bias, float* __restrict__ C,
    int M, int N, int K) {
  int lane = threadIdx.x & 31;
  int wave = threadIdx.x >> 5;
  int m0 = blockIdx.y * 64 + (wave & 3) * 16;
  int n0 = blockIdx.x * 32 + (wave >> 2) * 16;
  if (m0 >= M || n0 >= N) return;

  int am  = m0 + (lane & 15);        // A-frag row (lanes 0-15 / 16-31 both map m=lane&15)
  int akb = (lane >> 4) * 16;        // A-frag K sub-block
  int bn  = n0 + (lane & 15);        // B-frag column
  int bkb = (lane >> 4) * 16;        // B-frag K sub-block

  const _Float16* Arow = A  + (size_t)am * K + akb;
  const _Float16* Brow = Bt + (size_t)bn * K + bkb;

  v8f acc = {};
  for (int k0 = 0; k0 < K; k0 += 32) {
    v8h lo = *(const v8h*)(Arow + k0);
    v8h hi = *(const v8h*)(Arow + k0 + 8);
    v16h a = SHUF16(lo, hi);
    v16h b = *(const v16h*)(Brow + k0);
    acc = wmma_f16(a, b, acc);
  }
  float bb = bias ? bias[n0 + (lane & 15)] : 0.0f;
  int moff = (lane >> 4) * 8;
  int nn   = n0 + (lane & 15);
  for (int v = 0; v < 8; ++v)
    C[(size_t)(m0 + moff + v) * N + nn] = acc[v] + bb;
}

// ---------------------------------------------------------------------------
// GAT attention coefficients: als[n,h] = sum_c xl[n,h,c]*a_src[h,c]; same for ald.
// ---------------------------------------------------------------------------
__global__ void k_att_coef(const float* __restrict__ xl,
                           const float* __restrict__ asrc, const float* __restrict__ adst,
                           float* __restrict__ als, float* __restrict__ ald,
                           int H, int C) {
  int t = blockIdx.x * blockDim.x + threadIdx.x;
  if (t >= NN * H) return;
  int n = t / H, h = t - n * H;
  const float* xr = xl + (size_t)n * H * C + (size_t)h * C;
  const float* av = asrc + (size_t)h * C;
  const float* dv = adst + (size_t)h * C;
  float s = 0.f, d = 0.f;
  for (int c = 0; c < C; ++c) { float v = xr[c]; s += v * av[c]; d += v * dv[c]; }
  als[t] = s; ald[t] = d;
}

__global__ void k_init_seg(unsigned* __restrict__ segmax, float* __restrict__ denom, int n) {
  int t = blockIdx.x * blockDim.x + threadIdx.x;
  if (t < n) { segmax[t] = 0u; denom[t] = 0.f; }   // 0u <= ford_enc of any real float
}

static __device__ __forceinline__ void edge_sd(const long long* __restrict__ ei,
                                               int e, int& s, int& d) {
  if (e < NE) { s = (int)ei[e]; d = (int)ei[NE + e]; }
  else        { s = d = e - NE; }
}

// alpha = leaky_relu(als[src]+ald[dst], 0.2); segment max via ordered-uint atomicMax
__global__ void k_edge_alpha(const long long* __restrict__ ei,
                             const float* __restrict__ als, const float* __restrict__ ald,
                             float* __restrict__ ework, unsigned* __restrict__ segmax, int H) {
  int t = blockIdx.x * blockDim.x + threadIdx.x;
  if (t >= ETOT * H) return;
  int e = t / H, h = t - e * H;
  int s, d; edge_sd(ei, e, s, d);
  float a = als[s * H + h] + ald[d * H + h];
  a = (a > 0.f) ? a : 0.2f * a;
  ework[t] = a;
  atomicMax(&segmax[d * H + h], ford_enc(a));
}

// e = exp(alpha - max[dst]); denom[dst] += e
__global__ void k_edge_exp(const long long* __restrict__ ei,
                           float* __restrict__ ework, const unsigned* __restrict__ segmax,
                           float* __restrict__ denom, int H) {
  int t = blockIdx.x * blockDim.x + threadIdx.x;
  if (t >= ETOT * H) return;
  int e = t / H, h = t - e * H;
  int s, d; edge_sd(ei, e, s, d);
  float p = __expf(ework[t] - ford_dec(segmax[d * H + h]));
  ework[t] = p;
  atomicAdd(&denom[d * H + h], p);
}

// acc[dst, :] += (e/denom) * xl[src, :]   (one wave per edge)
__global__ __launch_bounds__(256) void k_edge_scatter(
    const long long* __restrict__ ei, const float* __restrict__ ework,
    const float* __restrict__ denom, const float* __restrict__ xl,
    float* __restrict__ acc, int H, int C) {
  int e = blockIdx.x * 8 + (threadIdx.x >> 5);
  if (e >= ETOT) return;
  int lane = threadIdx.x & 31;
  int s, d; edge_sd(ei, e, s, d);
  int HC = H * C;
  const float* xs = xl + (size_t)s * HC;
  float*       ad = acc + (size_t)d * HC;
  for (int c = lane; c < HC; c += 32) {
    int h = c / C;
    float w = ework[e * H + h] / (denom[d * H + h] + 1e-16f);
    atomicAdd(&ad[c], w * xs[c]);
  }
}

// ---------------------------------------------------------------------------
// out = LayerNorm( maybe_elu(acc + bias) ) * g + be  (+ res);  also write f16 copy.
// One wave per node; D in {256,512}.
// ---------------------------------------------------------------------------
__global__ __launch_bounds__(256) void k_bias_act_ln(
    const float* __restrict__ acc, const float* __restrict__ bias,
    const float* __restrict__ g, const float* __restrict__ be,
    const float* __restrict__ res,
    float* __restrict__ outf, _Float16* __restrict__ outh,
    int D, int do_elu) {
  int n = blockIdx.x * 8 + (threadIdx.x >> 5);
  if (n >= NN) return;
  int lane = threadIdx.x & 31;
  const float* xr = acc + (size_t)n * D;
  int cnt = D >> 5;                 // <= 16
  float vals[16];
  float s = 0.f;
  for (int i = 0; i < cnt; ++i) {
    int c = i * 32 + lane;
    float v = xr[c] + bias[c];
    if (do_elu) v = (v > 0.f) ? v : (__expf(v) - 1.f);
    vals[i] = v; s += v;
  }
  for (int off = 16; off; off >>= 1) s += __shfl_xor(s, off, 32);
  float mu = s / (float)D;
  float vs = 0.f;
  for (int i = 0; i < cnt; ++i) { float t = vals[i] - mu; vs += t * t; }
  for (int off = 16; off; off >>= 1) vs += __shfl_xor(vs, off, 32);
  float inv = rsqrtf(vs / (float)D + 1e-5f);
  for (int i = 0; i < cnt; ++i) {
    int c = i * 32 + lane;
    float y = (vals[i] - mu) * inv * g[c] + be[c];
    if (res) y += res[(size_t)n * D + c];
    outf[(size_t)n * D + c] = y;
    outh[(size_t)n * D + c] = (_Float16)y;
  }
}

// ---------------------------------------------------------------------------
// Repack projection outputs into head-major f16.
// mode 0: dst[h][n][dk]   (Q with 1/sqrt(dk) folded in, and K)
// mode 1: dst[h][dk][n]   (V transposed -> contiguous B operand for P*V)
// ---------------------------------------------------------------------------
__global__ void k_repack_head(const float* __restrict__ xl, _Float16* __restrict__ dst,
                              int mode, float scale) {
  int t = blockIdx.x * blockDim.x + threadIdx.x;
  if (t >= NN * 256) return;
  int n = t >> 8, c = t & 255;
  int h = c >> 5, dk = c & 31;
  _Float16 v = (_Float16)(xl[t] * scale);
  if (mode == 0) dst[((size_t)h * NN + n) * 32 + dk] = v;
  else           dst[((size_t)h * 32 + dk) * NN + n] = v;
}

// ---------------------------------------------------------------------------
// Flash attention: 8 heads, dk=32, seq=4096. One wave = one (head, 16-query tile).
// Scores via WMMA (Q[16x32] x K^T), online softmax with half-wave shuffles,
// P staged via LDS into A-fragment layout, P*V via WMMA with V^T operand.
// Output written as f16 [NN][256] (head-concat order) for the final projection.
// ---------------------------------------------------------------------------
__global__ __launch_bounds__(128) void k_flash_attn(
    const _Float16* __restrict__ Qh, const _Float16* __restrict__ Kh,
    const _Float16* __restrict__ Vt, _Float16* __restrict__ Oh) {
  __shared__ __align__(16) _Float16 pt[4][16 * 32];
  int wv = threadIdx.x >> 5, lane = threadIdx.x & 31;
  int gw = blockIdx.x * 4 + wv;                 // 0 .. 8*(NN/16)-1
  int h  = gw >> 8;                             // NN/16 == 256 tiles/head
  int q0 = (gw & 255) * 16;

  const _Float16* Q = Qh + (size_t)h * NN * 32;
  const _Float16* K = Kh + (size_t)h * NN * 32;
  const _Float16* V = Vt + (size_t)h * 32 * NN;

  int m   = lane & 15;
  int akb = (lane >> 4) * 16;
  int bn  = lane & 15;
  int bkb = (lane >> 4) * 16;

  v8h qlo = *(const v8h*)(Q + (size_t)(q0 + m) * 32 + akb);
  v8h qhi = *(const v8h*)(Q + (size_t)(q0 + m) * 32 + 8 + akb);
  v16h qa = SHUF16(qlo, qhi);

  v8f acc0 = {}, acc1 = {};
  float rm[8], rl[8];
  for (int v = 0; v < 8; ++v) { rm[v] = -1e30f; rl[v] = 0.f; }
  _Float16* myp = pt[wv];

  for (int kc = 0; kc < NN; kc += 32) {
    v16h kb0 = *(const v16h*)(K + (size_t)(kc + bn) * 32 + bkb);
    v16h kb1 = *(const v16h*)(K + (size_t)(kc + 16 + bn) * 32 + bkb);
    v8f z = {};
    v8f s0 = wmma_f16(qa, kb0, z);
    v8f s1 = wmma_f16(qa, kb1, z);

    for (int v = 0; v < 8; ++v) {
      float mx = fmaxf(s0[v], s1[v]);
      for (int off = 8; off; off >>= 1) mx = fmaxf(mx, __shfl_xor(mx, off, 32));
      float nm = fmaxf(rm[v], mx);
      float al = __expf(rm[v] - nm);
      float p0 = __expf(s0[v] - nm);
      float p1 = __expf(s1[v] - nm);
      float ps = p0 + p1;
      for (int off = 8; off; off >>= 1) ps += __shfl_xor(ps, off, 32);
      rl[v] = rl[v] * al + ps;
      rm[v] = nm;
      acc0[v] *= al; acc1[v] *= al;
      int row = v + ((lane >> 4) << 3);        // D-layout row for this lane half
      myp[row * 32 + (lane & 15)]      = (_Float16)p0;
      myp[row * 32 + 16 + (lane & 15)] = (_Float16)p1;
    }
    __syncthreads();
    v8h plo = *(const v8h*)(myp + m * 32 + akb);
    v8h phi = *(const v8h*)(myp + m * 32 + 8 + akb);
    v16h pa = SHUF16(plo, phi);
    __syncthreads();

    v16h vb0 = *(const v16h*)(V + (size_t)bn * NN + kc + bkb);
    v16h vb1 = *(const v16h*)(V + (size_t)(16 + bn) * NN + kc + bkb);
    acc0 = wmma_f16(pa, vb0, acc0);
    acc1 = wmma_f16(pa, vb1, acc1);
  }

  int moff = (lane >> 4) * 8;
  int nn = lane & 15;
  for (int v = 0; v < 8; ++v) {
    float invl = 1.f / rl[v];
    size_t q = (size_t)(q0 + moff + v);
    Oh[q * 256 + h * 32 + nn]      = (_Float16)(acc0[v] * invl);
    Oh[q * 256 + h * 32 + 16 + nn] = (_Float16)(acc1[v] * invl);
  }
}

// ---------------------------------------------------------------------------
// Host side
// ---------------------------------------------------------------------------
extern "C" void kernel_launch(void* const* d_in, const int* in_sizes, int n_in,
                              void* d_out, int out_size, void* d_ws, size_t ws_size,
                              hipStream_t stream) {
  (void)in_sizes; (void)n_in; (void)out_size; (void)ws_size;
  const float*     x   = (const float*)d_in[0];
  const long long* ei  = (const long long*)d_in[1];
  const float* W0  = (const float*)d_in[2];
  const float* as0 = (const float*)d_in[3];
  const float* ad0 = (const float*)d_in[4];
  const float* b0  = (const float*)d_in[5];
  const float* W1  = (const float*)d_in[6];
  const float* as1 = (const float*)d_in[7];
  const float* ad1 = (const float*)d_in[8];
  const float* b1  = (const float*)d_in[9];
  const float* W2  = (const float*)d_in[10];
  const float* as2 = (const float*)d_in[11];
  const float* ad2 = (const float*)d_in[12];
  const float* b2  = (const float*)d_in[13];
  const float* g0  = (const float*)d_in[14];
  const float* be0 = (const float*)d_in[15];
  const float* g1  = (const float*)d_in[16];
  const float* be1 = (const float*)d_in[17];
  const float* g2  = (const float*)d_in[18];
  const float* be2 = (const float*)d_in[19];
  const float* wq  = (const float*)d_in[20];
  const float* bq  = (const float*)d_in[21];
  const float* wk  = (const float*)d_in[22];
  const float* bk  = (const float*)d_in[23];
  const float* wv  = (const float*)d_in[24];
  const float* bv  = (const float*)d_in[25];
  const float* wo  = (const float*)d_in[26];
  const float* bo  = (const float*)d_in[27];

  // Workspace carve-up (all offsets 256B aligned)
  char* p = (char*)d_ws;
  _Float16* a16    = (_Float16*)p; p += (size_t)NN * 512 * 2;   // f16 GEMM input
  _Float16* w16    = (_Float16*)p; p += (size_t)512 * 512 * 2;  // transposed weights
  float*    xl     = (float*)p;    p += (size_t)NN * 512 * 4;   // GEMM out / proj out
  float*    acc    = (float*)p;    p += (size_t)NN * 512 * 4;   // GAT aggregation
  float*    hA     = (float*)p;    p += (size_t)NN * 512 * 4;   // node feats ping
  float*    hB     = (float*)p;    p += (size_t)NN * 512 * 4;   // node feats pong
  float*    als    = (float*)p;    p += (size_t)NN * 4 * 4;
  float*    ald    = (float*)p;    p += (size_t)NN * 4 * 4;
  unsigned* segmax = (unsigned*)p; p += (size_t)NN * 4 * 4;
  float*    denom  = (float*)p;    p += (size_t)NN * 4 * 4;
  float*    ework  = (float*)p;    p += (size_t)ETOT * 4 * 4;
  _Float16* qh     = (_Float16*)p; p += (size_t)8 * NN * 32 * 2;
  _Float16* kh     = (_Float16*)p; p += (size_t)8 * NN * 32 * 2;
  _Float16* vt     = (_Float16*)p; p += (size_t)8 * NN * 32 * 2;

  auto gemm = [&](const _Float16* A, const float* W, const float* bias,
                  float* C, int K, int Nc) {
    int tn = K * Nc;
    k_transpose_f16<<<(tn + 255) / 256, 256, 0, stream>>>(W, w16, K, Nc);
    k_gemm_wmma<<<dim3(Nc / 32, NN / 64), 256, 0, stream>>>(A, w16, bias, C, NN, Nc, K);
  };

  auto gat_layer = [&](const float* W, const float* as_, const float* ad_,
                       const float* b_, const float* g_, const float* be_,
                       const float* res, float* outf,
                       int Kin, int H, int C, int do_elu) {
    gemm(a16, W, nullptr, xl, Kin, H * C);
    k_att_coef<<<(NN * H + 255) / 256, 256, 0, stream>>>(xl, as_, ad_, als, ald, H, C);
    k_init_seg<<<(NN * H + 255) / 256, 256, 0, stream>>>(segmax, denom, NN * H);
    hipMemsetAsync(acc, 0, (size_t)NN * H * C * 4, stream);
    k_edge_alpha<<<(ETOT * H + 255) / 256, 256, 0, stream>>>(ei, als, ald, ework, segmax, H);
    k_edge_exp<<<(ETOT * H + 255) / 256, 256, 0, stream>>>(ei, ework, segmax, denom, H);
    k_edge_scatter<<<(ETOT + 7) / 8, 256, 0, stream>>>(ei, ework, denom, xl, acc, H, C);
    k_bias_act_ln<<<(NN + 7) / 8, 256, 0, stream>>>(acc, b_, g_, be_, res,
                                                    outf, a16, H * C, do_elu);
  };

  // input features -> f16
  k_to_f16<<<(NN * 128 + 255) / 256, 256, 0, stream>>>(x, a16, NN * 128);

  // GAT layer 0: 128 -> 4x128 (elu + LN)
  gat_layer(W0, as0, ad0, b0, g0, be0, nullptr, hA, 128, 4, 128, 1);
  // GAT layer 1: 512 -> 4x128 (elu + LN + residual)
  gat_layer(W1, as1, ad1, b1, g1, be1, hA, hB, 512, 4, 128, 1);
  // GAT layer 2: 512 -> 1x256 (LN only)
  gat_layer(W2, as2, ad2, b2, g2, be2, nullptr, hA, 512, 1, 256, 0);

  // Dense MHA: projections (a16 holds f16 LN2 output, [NN][256])
  const float qscale = 0.17677669529663687f;   // 1/sqrt(32)
  gemm(a16, wq, bq, xl, 256, 256);
  k_repack_head<<<(NN * 256 + 255) / 256, 256, 0, stream>>>(xl, qh, 0, qscale);
  gemm(a16, wk, bk, xl, 256, 256);
  k_repack_head<<<(NN * 256 + 255) / 256, 256, 0, stream>>>(xl, kh, 0, 1.0f);
  gemm(a16, wv, bv, xl, 256, 256);
  k_repack_head<<<(NN * 256 + 255) / 256, 256, 0, stream>>>(xl, vt, 1, 1.0f);

  // flash attention -> a16 (f16 [NN][256], head-concat columns)
  k_flash_attn<<<(8 * (NN / 16)) / 4, 128, 0, stream>>>(qh, kh, vt, a16);

  // output projection -> d_out (f32 [NN][256])
  gemm(a16, wo, bo, (float*)d_out, 256, 256);
}
